// MoeHashLayer_16922171146615
// MI455X (gfx1250) — compile-verified, matrix operand order checked by reading, not run on hardware
//
#include <hip/hip_runtime.h>
#include <hip/hip_bf16.h>

// ---------------------------------------------------------------------------
// MoE hash layer for MI455X (gfx1250, wave32, WMMA).
// Token-grouped bf16 WMMA GEMMs with double-buffered LDS + register prefetch.
// GEMM2's A tile (already bf16 in workspace) is staged with the CDNA5 async
// global->LDS engine (GLOBAL_LOAD_ASYNC_TO_LDS_B128, ASYNCcnt), so it never
// touches VGPRs; f32 weights still stage through VGPRs for bf16 conversion.
// ---------------------------------------------------------------------------

#define NTOK   8192      // BSZ*CNTX
#define DIMC   1024
#define HIDC   4096
#define NEXP   8
#define MT     128       // M tile (tokens)
#define NT     128       // N tile
#define KT     32        // K step (bf16 WMMA K)
#define MT_MAX (NTOK / MT)   // 64 worst-case M tiles per expert

typedef unsigned int u32;
typedef __bf16 bf16_t;
typedef __attribute__((ext_vector_type(16))) __bf16 v16bf;
typedef __attribute__((ext_vector_type(8)))  float  v8f;
typedef __attribute__((ext_vector_type(4)))  float  v4f;
typedef __attribute__((ext_vector_type(4)))  u32    v4u;
typedef __attribute__((ext_vector_type(2)))  u32    v2u;

union Frag { v4u q[2]; v16bf v; };   // 32B: one WMMA A/B operand (8 VGPRs)

__device__ __forceinline__ unsigned short bf_bits(float f) {
  u32 u = __builtin_bit_cast(u32, f);
  u32 r = u + 0x7FFFu + ((u >> 16) & 1u);          // round-to-nearest-even
  return (unsigned short)(r >> 16);
}
__device__ __forceinline__ bf16_t tobf(float f) {
  unsigned short s = bf_bits(f);
  return __builtin_bit_cast(bf16_t, s);
}
// pack 4 floats -> 4 bf16, 8-byte LDS store
__device__ __forceinline__ void st4bf(bf16_t* p, v4f v) {
  v2u u;
  u.x = (u32)bf_bits(v.x) | ((u32)bf_bits(v.y) << 16);
  u.y = (u32)bf_bits(v.z) | ((u32)bf_bits(v.w) << 16);
  *(v2u*)p = u;
}

#define WMMA_BF16(A, B, C) \
  __builtin_amdgcn_wmma_f32_16x16x32_bf16(false, (A), false, (B), (short)0, (C), false, false)

// async global -> LDS copy of 32B/lane (2 x b128; offset applies to both sides)
__device__ __forceinline__ void async_copy32(const bf16_t* gsrc, bf16_t* ldst) {
  u32 lo = (u32)(uintptr_t)ldst;                       // LDS byte address = addr[31:0]
  unsigned long long ga = (unsigned long long)(uintptr_t)gsrc;
  asm volatile("global_load_async_to_lds_b128 %0, %1, off\n\t"
               "global_load_async_to_lds_b128 %0, %1, off offset:16"
               :: "v"(lo), "v"(ga) : "memory");
}
__device__ __forceinline__ void wait_async0() {
  asm volatile("s_wait_asynccnt 0x0" ::: "memory");
}

// ---------------------------------------------------------------------------
// Phase 0: group tokens by expert (histogram -> prefix -> scatter)
// ---------------------------------------------------------------------------
__global__ void k_init(int* counts, int* cursors) {
  int t = threadIdx.x;
  if (t < NEXP) { counts[t] = 0; cursors[t] = 0; }
}
__global__ void k_hist(const int* __restrict__ maps, int* __restrict__ counts) {
  int t = blockIdx.x * blockDim.x + threadIdx.x;
  if (t < NTOK) atomicAdd(&counts[maps[t]], 1);
}
__global__ void k_scan(const int* __restrict__ counts, int* __restrict__ offs) {
  if (threadIdx.x == 0) {
    int s = 0;
    for (int i = 0; i < NEXP; ++i) { offs[i] = s; s += counts[i]; }
    offs[NEXP] = s;
  }
}
__global__ void k_scatter(const int* __restrict__ maps, const int* __restrict__ offs,
                          int* __restrict__ cursors, int* __restrict__ perm) {
  int t = blockIdx.x * blockDim.x + threadIdx.x;
  if (t < NTOK) {
    int e = maps[t];
    int p = atomicAdd(&cursors[e], 1);
    perm[offs[e] + p] = t;
  }
}

// ---------------------------------------------------------------------------
// GEMM1 helpers
// ---------------------------------------------------------------------------
__device__ __forceinline__ void g1_load(const float* __restrict__ xrow,
                                        const float* __restrict__ wgp,
                                        const float* __restrict__ wip, int k0,
                                        v4f (&ar)[4], v4f (&gr)[4], v4f (&ir)[4]) {
  const v4f* xa = (const v4f*)(xrow + k0);
  const v4f* g4 = (const v4f*)(wgp + (size_t)k0 * HIDC);
  const v4f* i4 = (const v4f*)(wip + (size_t)k0 * HIDC);
#pragma unroll
  for (int c = 0; c < 4; ++c) { ar[c] = xa[c]; gr[c] = g4[c]; ir[c] = i4[c]; }
}

__device__ __forceinline__ void g1_stage(bf16_t (*As)[40], bf16_t (*Bgs)[40], bf16_t (*Bis)[40],
                                         int a_m, int a_kh, int b_k, int b_n,
                                         const v4f (&ar)[4], const v4f (&gr)[4], const v4f (&ir)[4]) {
  bf16_t* ap = &As[a_m][a_kh];
#pragma unroll
  for (int c = 0; c < 4; ++c) st4bf(ap + 4 * c, ar[c]);
#pragma unroll
  for (int c = 0; c < 4; ++c) {
    int n = b_n + c * 4;
    Bgs[n + 0][b_k] = tobf(gr[c].x); Bgs[n + 1][b_k] = tobf(gr[c].y);
    Bgs[n + 2][b_k] = tobf(gr[c].z); Bgs[n + 3][b_k] = tobf(gr[c].w);
    Bis[n + 0][b_k] = tobf(ir[c].x); Bis[n + 1][b_k] = tobf(ir[c].y);
    Bis[n + 2][b_k] = tobf(ir[c].z); Bis[n + 3][b_k] = tobf(ir[c].w);
  }
}

// ---------------------------------------------------------------------------
// GEMM1: h[p, :] = silu(x_g @ Wg + bg) * (x_g @ Wi + bi)   (bf16 out, permuted)
// block = 128 tokens x 128 hid cols, 256 threads (8 waves), ping-pong LDS
// ---------------------------------------------------------------------------
__global__ __launch_bounds__(256)
void moe_gemm1(const float* __restrict__ x,
               const float* __restrict__ Wg, const float* __restrict__ bg,
               const float* __restrict__ Wi, const float* __restrict__ bi,
               const int* __restrict__ offs, const int* __restrict__ perm,
               bf16_t* __restrict__ h)
{
  __shared__ bf16_t As [2][MT][40];   // ping-pong 128 x 32 (pad 40)
  __shared__ bf16_t Bgs[2][NT][40];   // transposed: [n][k]
  __shared__ bf16_t Bis[2][NT][40];   // total 60 KB

  const int e      = blockIdx.x / MT_MAX;
  const int mt     = blockIdx.x % MT_MAX;
  const int row0   = offs[e] + mt * MT;
  const int rowEnd = offs[e + 1];
  if (row0 >= rowEnd) return;                    // uniform early-out (dead tile)
  const int rows = rowEnd - row0;                // effective rows = min(rows, MT)
  const int nc   = blockIdx.y * NT;

  const int tid  = threadIdx.x;
  const int lane = tid & 31;
  const int wave = tid >> 5;
  const int wrow = (wave & 3) * 32;              // wave's 32-row patch
  const int wcol = (wave >> 2) * 64;             // wave's 64-col patch

  // A gather coords: thread loads 16 floats of one token row
  const int a_m  = tid >> 1;
  const int a_kh = (tid & 1) * 16;
  const int a_row = (a_m < rows) ? a_m : (rows - 1);
  const int tokA  = perm[row0 + a_row];
  const float* xrow = x + (size_t)tokA * DIMC + a_kh;

  // B coords: thread loads 16 cols of one K row from each weight
  const int b_k = tid >> 3;
  const int b_n = (tid & 7) * 16;
  const float* wgp = Wg + (size_t)e * DIMC * HIDC + (size_t)b_k * HIDC + nc + b_n;
  const float* wip = Wi + (size_t)e * DIMC * HIDC + (size_t)b_k * HIDC + nc + b_n;

  v8f accg[2][4], acci[2][4];
#pragma unroll
  for (int i = 0; i < 2; ++i)
#pragma unroll
    for (int j = 0; j < 4; ++j) { accg[i][j] = (v8f)0.0f; acci[i][j] = (v8f)0.0f; }

  // prologue: fetch + stage K-tile 0 into buffer 0
  v4f ar[4], gr[4], ir[4];
  g1_load(xrow, wgp, wip, 0, ar, gr, ir);
  g1_stage(As[0], Bgs[0], Bis[0], a_m, a_kh, b_k, b_n, ar, gr, ir);

  const int fl = lane & 15;
  const int kh = (lane >> 4) * 8;                // K-half per 16-bit operand layout

  int buf = 0;
  for (int k0 = 0; k0 < DIMC; k0 += KT) {
    __syncthreads();
    const bool hasNext = (k0 + KT) < DIMC;
    if (hasNext) g1_load(xrow, wgp, wip, k0 + KT, ar, gr, ir);   // prefetch next tile

    bf16_t (*cAs)[40]  = As[buf];
    bf16_t (*cBgs)[40] = Bgs[buf];
    bf16_t (*cBis)[40] = Bis[buf];

    Frag a0, a1;
    a0.q[0] = *(const v4u*)&cAs[wrow + fl][kh];
    a0.q[1] = *(const v4u*)&cAs[wrow + fl][16 + kh];
    a1.q[0] = *(const v4u*)&cAs[wrow + 16 + fl][kh];
    a1.q[1] = *(const v4u*)&cAs[wrow + 16 + fl][16 + kh];
#pragma unroll
    for (int j = 0; j < 4; ++j) {
      const int ncl = wcol + j * 16 + fl;
      Frag bgf, bif;
      bgf.q[0] = *(const v4u*)&cBgs[ncl][kh];
      bgf.q[1] = *(const v4u*)&cBgs[ncl][16 + kh];
      bif.q[0] = *(const v4u*)&cBis[ncl][kh];
      bif.q[1] = *(const v4u*)&cBis[ncl][16 + kh];
      accg[0][j] = WMMA_BF16(a0.v, bgf.v, accg[0][j]);
      accg[1][j] = WMMA_BF16(a1.v, bgf.v, accg[1][j]);
      acci[0][j] = WMMA_BF16(a0.v, bif.v, acci[0][j]);
      acci[1][j] = WMMA_BF16(a1.v, bif.v, acci[1][j]);
    }

    if (hasNext)   // stage next tile into the other buffer (reads of it ended
                   // before this iteration's barrier)
      g1_stage(As[buf ^ 1], Bgs[buf ^ 1], Bis[buf ^ 1], a_m, a_kh, b_k, b_n, ar, gr, ir);
    buf ^= 1;
  }

  // --- epilogue: bias + silu(gate)*in -> h (bf16, permuted row order) ---
  const int rb = (lane >> 4) * 8;                // C-matrix row base per lane half
#pragma unroll
  for (int i = 0; i < 2; ++i) {
#pragma unroll
    for (int j = 0; j < 4; ++j) {
      const int ng  = nc + wcol + j * 16 + fl;
      const float vbg = bg[e * HIDC + ng];
      const float vbi = bi[e * HIDC + ng];
#pragma unroll
      for (int r = 0; r < 8; ++r) {
        const int m = wrow + i * 16 + rb + r;
        if (m < rows) {
          float g  = accg[i][j][r] + vbg;
          float iv = acci[i][j][r] + vbi;
          float s  = g / (1.0f + __expf(-g));    // silu
          h[(size_t)(row0 + m) * HIDC + ng] = tobf(s * iv);
        }
      }
    }
  }
}

// ---------------------------------------------------------------------------
// GEMM2 helpers (A staged by async global->LDS; B through VGPRs for cvt)
// ---------------------------------------------------------------------------
__device__ __forceinline__ void g2_loadB(const float* __restrict__ wop, int k0,
                                         v4f (&orr)[4]) {
  const v4f* o4 = (const v4f*)(wop + (size_t)k0 * DIMC);
#pragma unroll
  for (int c = 0; c < 4; ++c) orr[c] = o4[c];
}

__device__ __forceinline__ void g2_stageB(bf16_t (*Bs)[40], int b_k, int b_n,
                                          const v4f (&orr)[4]) {
#pragma unroll
  for (int c = 0; c < 4; ++c) {
    int n = b_n + c * 4;
    Bs[n + 0][b_k] = tobf(orr[c].x); Bs[n + 1][b_k] = tobf(orr[c].y);
    Bs[n + 2][b_k] = tobf(orr[c].z); Bs[n + 3][b_k] = tobf(orr[c].w);
  }
}

// ---------------------------------------------------------------------------
// GEMM2: out[tok, :] = h[p, :] @ Wo[e] + bo[e]   (scatter through perm, f32)
// ---------------------------------------------------------------------------
__global__ __launch_bounds__(256)
void moe_gemm2(const bf16_t* __restrict__ h,
               const float* __restrict__ Wo, const float* __restrict__ bo,
               const int* __restrict__ offs, const int* __restrict__ perm,
               float* __restrict__ out)
{
  __shared__ bf16_t As[2][MT][40];
  __shared__ bf16_t Bs[2][NT][40];   // total 40 KB

  const int e      = blockIdx.x / MT_MAX;
  const int mt     = blockIdx.x % MT_MAX;
  const int row0   = offs[e] + mt * MT;
  const int rowEnd = offs[e + 1];
  if (row0 >= rowEnd) return;
  const int rows = rowEnd - row0;
  const int nc   = blockIdx.y * NT;

  const int tid  = threadIdx.x;
  const int lane = tid & 31;
  const int wave = tid >> 5;
  const int wrow = (wave & 3) * 32;
  const int wcol = (wave >> 2) * 64;

  const int a_m  = tid >> 1;
  const int a_kh = (tid & 1) * 16;
  const int a_row = (a_m < rows) ? a_m : (rows - 1);
  const bf16_t* hrow = h + (size_t)(row0 + a_row) * HIDC + a_kh;

  const int b_k = tid >> 3;
  const int b_n = (tid & 7) * 16;
  const float* wop = Wo + (size_t)e * HIDC * DIMC + (size_t)b_k * DIMC + nc + b_n;

  v8f acc[2][4];
#pragma unroll
  for (int i = 0; i < 2; ++i)
#pragma unroll
    for (int j = 0; j < 4; ++j) acc[i][j] = (v8f)0.0f;

  // prologue: async A tile 0 -> LDS, B tile 0 via VGPRs
  v4f orr[4];
  async_copy32(hrow, &As[0][a_m][a_kh]);
  g2_loadB(wop, 0, orr);
  g2_stageB(Bs[0], b_k, b_n, orr);

  const int fl = lane & 15;
  const int kh = (lane >> 4) * 8;

  int buf = 0;
  for (int k0 = 0; k0 < HIDC; k0 += KT) {
    wait_async0();                       // A tile for this iteration landed
    __syncthreads();
    const bool hasNext = (k0 + KT) < HIDC;
    if (hasNext) {
      async_copy32(hrow + (k0 + KT), &As[buf ^ 1][a_m][a_kh]);  // cache->LDS, no VGPRs
      g2_loadB(wop, k0 + KT, orr);                              // prefetch B regs
    }

    bf16_t (*cAs)[40] = As[buf];
    bf16_t (*cBs)[40] = Bs[buf];

    Frag a0, a1;
    a0.q[0] = *(const v4u*)&cAs[wrow + fl][kh];
    a0.q[1] = *(const v4u*)&cAs[wrow + fl][16 + kh];
    a1.q[0] = *(const v4u*)&cAs[wrow + 16 + fl][kh];
    a1.q[1] = *(const v4u*)&cAs[wrow + 16 + fl][16 + kh];
#pragma unroll
    for (int j = 0; j < 4; ++j) {
      const int ncl = wcol + j * 16 + fl;
      Frag bf;
      bf.q[0] = *(const v4u*)&cBs[ncl][kh];
      bf.q[1] = *(const v4u*)&cBs[ncl][16 + kh];
      acc[0][j] = WMMA_BF16(a0.v, bf.v, acc[0][j]);
      acc[1][j] = WMMA_BF16(a1.v, bf.v, acc[1][j]);
    }

    if (hasNext)
      g2_stageB(Bs[buf ^ 1], b_k, b_n, orr);
    buf ^= 1;
  }

  const int rb = (lane >> 4) * 8;
#pragma unroll
  for (int i = 0; i < 2; ++i) {
#pragma unroll
    for (int j = 0; j < 4; ++j) {
      const int ng  = nc + wcol + j * 16 + fl;
      const float vb = bo[e * DIMC + ng];
#pragma unroll
      for (int r = 0; r < 8; ++r) {
        const int m = wrow + i * 16 + rb + r;
        if (m < rows) {
          const int tok = perm[row0 + m];
          out[(size_t)tok * DIMC + ng] = acc[i][j][r] + vb;
        }
      }
    }
  }
}

// ---------------------------------------------------------------------------
extern "C" void kernel_launch(void* const* d_in, const int* in_sizes, int n_in,
                              void* d_out, int out_size, void* d_ws, size_t ws_size,
                              hipStream_t stream) {
  (void)in_sizes; (void)n_in; (void)out_size; (void)ws_size;
  const float* x    = (const float*)d_in[0];
  const int*   maps = (const int*)  d_in[1];
  const float* W_in = (const float*)d_in[2];
  const float* b_in = (const float*)d_in[3];
  const float* W_g  = (const float*)d_in[4];
  const float* b_g  = (const float*)d_in[5];
  const float* W_o  = (const float*)d_in[6];
  const float* b_o  = (const float*)d_in[7];
  float* out = (float*)d_out;

  // workspace layout
  int* counts  = (int*)d_ws;          // [8]
  int* offs    = counts + 8;          // [9]
  int* cursors = counts + 17;         // [8]
  int* perm    = counts + 32;         // [8192]
  bf16_t* hws  = (bf16_t*)((char*)d_ws + (1 << 16));  // 8192 x 4096 bf16 = 64MB

  k_init<<<1, 32, 0, stream>>>(counts, cursors);
  k_hist<<<NTOK / 256, 256, 0, stream>>>(maps, counts);
  k_scan<<<1, 1, 0, stream>>>(counts, offs);
  k_scatter<<<NTOK / 256, 256, 0, stream>>>(maps, offs, cursors, perm);

  dim3 g1(NEXP * MT_MAX, HIDC / NT);   // (512, 32)
  moe_gemm1<<<g1, 256, 0, stream>>>(x, W_g, b_g, W_in, b_in, offs, perm, hws);

  dim3 g2(NEXP * MT_MAX, DIMC / NT);   // (512, 8)
  moe_gemm2<<<g2, 256, 0, stream>>>(hws, W_o, b_o, offs, perm, out);
}
